// DCNN_47253230191005
// MI455X (gfx1250) — matrix operand order repaired and checked by reading, not compiled, not run on hardware
//
#include <hip/hip_runtime.h>
#include <hip/hip_bf16.h>
#include <math.h>

// ---------------------------------------------------------------------------
// DCNN forward, fused for MI455X (gfx1250, wave32).
//   K1 = conv1 + dynamic k-max(K=256) + tanh   -> P1 (f16)   [B,150,6,256]
//   K2 = TDM-staged conv2 (WMMA f16 16x16x32) + k-max(4) + tanh
//        -> pooled2 (f32) [B,4,1050]
//   K3 = FC (GEMV) -> out [B,2]
// ---------------------------------------------------------------------------

#define TT     512      // time steps
#define CC     300      // channels
#define NG1    150      // conv1 groups (C/2)
#define NG2    75       // conv2 groups (C/4)
#define K1MAX  256      // static K for layer-1 k-max: max(4,(T+1)/2)
#define FEAT2  1050     // 14 * NG2
#define FCDIM  4200     // 4 * FEAT2

typedef __attribute__((ext_vector_type(16))) _Float16 v16h;
typedef __attribute__((ext_vector_type(8)))  _Float16 v8h;
typedef __attribute__((ext_vector_type(8)))  float    v8f;
typedef __attribute__((ext_vector_type(4)))  unsigned v4u;
typedef __attribute__((ext_vector_type(8)))  int      v8i;
typedef __attribute__((ext_vector_type(4)))  int      v4i;

__device__ __forceinline__ unsigned order_key(float f) {
    unsigned u = __float_as_uint(f);
    return u ^ ((u & 0x80000000u) ? 0xFFFFFFFFu : 0x80000000u);
}

// ---------------------------------------------------------------------------
// Kernel 1: one block (6 waves) per (b, g).  Wave j owns out-channel 6g+j.
// ---------------------------------------------------------------------------
__global__ void __launch_bounds__(192)
dcnn_conv1_kmax(const float* __restrict__ x, const int* __restrict__ lengths,
                const float* __restrict__ w1, const float* __restrict__ b1,
                _Float16* __restrict__ P1) {
    const int b    = blockIdx.x / NG1;
    const int g    = blockIdx.x % NG1;
    const int lane = threadIdx.x & 31;
    const int wv   = threadIdx.x >> 5;          // 0..5  (= j)
    const int len  = lengths[b];
    const int PR   = len + 6;                   // valid prefix of conv1 output
    int k1 = (len + 1) >> 1; if (k1 < 4) k1 = 4;

    __shared__ float xsh[2][528];               // padded so tin<528 always safe

    // the two group channels are adjacent -> one b64 load per time step
    for (int t = threadIdx.x; t < 528; t += blockDim.x) {
        float2 v = make_float2(0.f, 0.f);
        if (t < len)
            v = *(const float2*)&x[(size_t)b * TT * CC + (size_t)t * CC + 2 * g];
        xsh[0][t] = v.x;
        xsh[1][t] = v.y;
    }
    // zero-fill this block's P1 region (scratch is poisoned by the harness)
    _Float16* pbase = P1 + (size_t)(b * NG1 + g) * 6 * K1MAX;
    for (int idx = threadIdx.x; idx < 6 * K1MAX; idx += blockDim.x)
        pbase[idx] = (_Float16)0.f;
    __syncthreads();

    const int oc = 6 * g + wv;
    float wr[2][7];
    #pragma unroll
    for (int ic = 0; ic < 2; ++ic)
        #pragma unroll
        for (int k = 0; k < 7; ++k) wr[ic][k] = w1[(oc * 2 + ic) * 7 + k];
    const float bias = b1[oc];

    // conv1: lane owns 17 contiguous time steps (17*32 = 544 >= PR)
    float    hv[17];
    unsigned key[17];
    const int t0 = lane * 17;
    #pragma unroll
    for (int i = 0; i < 17; ++i) {
        const int t = t0 + i;
        if (t < PR) {
            float acc = bias;
            #pragma unroll
            for (int ic = 0; ic < 2; ++ic)
                #pragma unroll
                for (int k = 0; k < 7; ++k) {
                    const int tin = t + k - 6;
                    if (tin >= 0) acc += xsh[ic][tin] * wr[ic][k];
                }
            hv[i]  = acc;
            key[i] = order_key(acc);
        } else { hv[i] = 0.f; key[i] = 0u; }
    }

    // binary search the k1-th largest key across the wave
    unsigned lo = 0u, hi = 0xFFFFFFFFu;
    while (lo < hi) {
        const unsigned d   = hi - lo;
        const unsigned mid = lo + (d >> 1) + (d & 1u);   // ceil midpoint
        int cnt = 0;
        #pragma unroll
        for (int i = 0; i < 17; ++i) cnt += (key[i] >= mid) ? 1 : 0;
        for (int off = 16; off; off >>= 1) cnt += __shfl_xor(cnt, off, 32);
        if (cnt >= k1) lo = mid; else hi = mid - 1;
    }
    const unsigned thr = lo;

    // wave prefix sums of (>thr) and (==thr) counts -> temporal-order emission
    int gtl = 0, eql = 0;
    #pragma unroll
    for (int i = 0; i < 17; ++i) {
        gtl += (key[i] >  thr) ? 1 : 0;
        eql += (key[i] == thr) ? 1 : 0;
    }
    int gts = gtl, eqs = eql;
    for (int off = 1; off < 32; off <<= 1) {
        const int vg = __shfl_up(gts, off, 32);
        const int ve = __shfl_up(eqs, off, 32);
        if (lane >= off) { gts += vg; eqs += ve; }
    }
    const int totalGT = __shfl(gts, 31, 32);
    const int needE   = k1 - totalGT;           // ties kept (earliest first)
    int g_run = gts - gtl;
    int e_run = eqs - eql;
    _Float16* prow = pbase + wv * K1MAX;
    #pragma unroll
    for (int i = 0; i < 17; ++i) {
        const int t = t0 + i;
        if (t < PR) {
            if (key[i] > thr) {
                const int ekeep = (e_run < needE) ? e_run : needE;
                prow[g_run + ekeep] = (_Float16)tanhf(hv[i]);
                ++g_run;
            } else if (key[i] == thr) {
                if (e_run < needE)
                    prow[g_run + e_run] = (_Float16)tanhf(hv[i]);
                ++e_run;
            }
        }
    }
}

// ---------------------------------------------------------------------------
// Kernel 2: one wave per (b, gg).
//   - P1 tile (12 rows x 256 f16) DMA'd into padded LDS rows by the TDM,
//     halo realized via D# pad_interval/pad_amount (288-half row pitch).
//   - conv2 as [260 x 60] x [60 x 14] GEMM via v_wmma_f32_16x16x32_f16
//     (17 M-tiles x 2 K-chunks), fragments from LDS as aligned b128 loads.
//   - fused top-4 k-max + tanh.
// ---------------------------------------------------------------------------
__global__ void __launch_bounds__(32)
dcnn_conv2_kmax_wmma(const _Float16* __restrict__ P1,
                     const int* __restrict__ lengths,
                     const float* __restrict__ w2, const float* __restrict__ b2,
                     float* __restrict__ pooled2) {
    const int b    = blockIdx.x / NG2;
    const int gg   = blockIdx.x % NG2;
    const int lane = threadIdx.x;               // 0..31

    __shared__ __align__(16) _Float16 shIn[12][288]; // [2*6 rows][4 | 256 | 28]
    __shared__ __align__(64) _Float16 shA[272][64];  // im2col, K padded 60->64
    __shared__ __align__(64) _Float16 shBt[16][64];  // weights, n-major
    __shared__ __align__(16) float    shOut[272][16];

    // zero only the halo columns of shIn (disjoint from the TDM footprint)
    for (int idx = lane; idx < 12 * 32; idx += 32) {
        const int row = idx >> 5, c = idx & 31;
        const int col = (c < 4) ? c : (256 + c);     // [0,4) and [260,288)
        shIn[row][col] = (_Float16)0.f;
    }

    // --- Tensor Data Mover: P1[b, 2gg.., :, :] -> shIn rows at +4 halfs ----
    {
        const _Float16* src = P1 + (size_t)(b * NG1 + 2 * gg) * 6 * K1MAX;
        const unsigned long long ga = (unsigned long long)(size_t)src;
        const unsigned lds_base = (unsigned)(size_t)&shIn[0][4];
        v4u g0;
        g0[0] = 1u;                                  // count=1 (user D#)
        g0[1] = lds_base;                            // lds_addr
        g0[2] = (unsigned)(ga & 0xFFFFFFFFull);      // global_addr[31:0]
        g0[3] = (unsigned)((ga >> 32) & 0x01FFFFFFull) | (2u << 30); // type=2
        v8i g1;
        g1[0] = (int)((1u << 16)      // data_size = 2 bytes
                    | (1u << 20)      // pad_enable
                    | (6u << 22)      // pad_interval: every 128 DWORDs (512B)
                    | (15u << 25));   // pad_amount: 16 DWORDs (64B) -> 576B pitch
        g1[1] = (int)(256u << 16);    // tensor_dim0 = 256  (bits 79:48)
        g1[2] = (int)(12u  << 16);    // tensor_dim1 = 12   (bits 111:80)
        g1[3] = (int)(256u << 16);    // tile_dim0   = 256  (bits 127:112)
        g1[4] = 12;                   // tile_dim1   = 12   (bits 143:128)
        g1[5] = 256;                  // tensor_dim0_stride = 256
        g1[6] = 0;
        g1[7] = 0;
        const v4i gz4 = {0, 0, 0, 0};
        const v8i gz8 = {0, 0, 0, 0, 0, 0, 0, 0};
        // 6-arg form (clang-23 / therock-10.0 toolchain)
        __builtin_amdgcn_tensor_load_to_lds(g0, g1, gz4, gz4, gz8, 0);
    }

    // stage weights n-major with zero padding (overlaps the TDM)
    for (int idx = lane; idx < 16 * 64; idx += 32) {
        const int n = idx >> 6, K = idx & 63;
        _Float16 v = (_Float16)0.f;
        if (n < 14 && K < 60) v = (_Float16)w2[(14 * gg + n) * 60 + K];
        shBt[n][K] = v;
    }

    __builtin_amdgcn_s_wait_tensorcnt(0);
    __syncthreads();

    // build im2col: shA[w][K] = shIn[i*6 + kh][w + kw], K = i*30 + kh*5 + kw
    for (int idx = lane; idx < 272 * 64; idx += 32) {
        const int w = idx >> 6, K = idx & 63;
        _Float16 v = (_Float16)0.f;
        if (K < 60) {
            const int i  = K / 30;
            const int r  = K % 30;
            const int kh = r / 5;
            const int kw = r % 5;
            v = shIn[i * 6 + kh][w + kw];   // pad -4 baked into +4 base offset
        }
        shA[w][K] = v;
    }
    __syncthreads();

    const int m_a  = lane & 15;                 // A fragment: output row w-offset
    const int hs   = (lane >> 4) * 16;          // A K-base per lane half
    const int kb   = (lane >> 4) * 16;          // B K-base per lane half
    const int n    = lane & 15;                 // B/D column (out channel)
    const int mrow = (lane >> 4) * 8;           // D row base per lane half
    const float bias = (n < 14) ? b2[14 * gg + n] : 0.f;

    // B fragments are tile-invariant: hoist both K-chunks into registers
    const v16h bf0 = *(const v16h*)&shBt[n][kb];        // K = kb + e
    const v16h bf1 = *(const v16h*)&shBt[n][32 + kb];   // K = 32 + kb + e

    #pragma unroll
    for (int tile = 0; tile < 17; ++tile) {
        const int w0 = tile * 16;
        const _Float16* arow = &shA[w0 + m_a][0];
        // A(16x32) contiguous: K = hs + e, matching B
        const v8h a0 = *(const v8h*)(arow + hs);
        const v8h a1 = *(const v8h*)(arow + 8 + hs);
        const v8h a2 = *(const v8h*)(arow + 32 + hs);
        const v8h a3 = *(const v8h*)(arow + 40 + hs);
        const v16h A0 = __builtin_shufflevector(a0, a1,
            0, 1, 2, 3, 4, 5, 6, 7, 8, 9, 10, 11, 12, 13, 14, 15);
        const v16h A1 = __builtin_shufflevector(a2, a3,
            0, 1, 2, 3, 4, 5, 6, 7, 8, 9, 10, 11, 12, 13, 14, 15);
        v8f c = {};
        c = __builtin_amdgcn_wmma_f32_16x16x32_f16(
                false, A0, false, bf0, (short)0, c, false, false);
        c = __builtin_amdgcn_wmma_f32_16x16x32_f16(
                false, A1, false, bf1, (short)0, c, false, false);
        #pragma unroll
        for (int r = 0; r < 8; ++r)
            shOut[w0 + mrow + r][n] = c[r] + bias;   // D: M = r + mrow, N = n
    }
    __syncthreads();

    // layer-2 dynamic k-max: top-4 over w in [0, k1+4), order preserved
    if (lane < 14) {
        const int len = lengths[b];
        int k1 = (len + 1) >> 1; if (k1 < 4) k1 = 4;
        const int Wd = k1 + 4;                   // <= 260, >= 8
        int   pick[4]; float pval[4];
        #pragma unroll
        for (int s = 0; s < 4; ++s) { pick[s] = -1; pval[s] = 0.f; }
        for (int s = 0; s < 4; ++s) {
            float best = -3.4e38f; int bw = 0;
            for (int w = 0; w < Wd; ++w) {
                bool skip = false;
                for (int q = 0; q < s; ++q) skip = skip || (pick[q] == w);
                if (skip) continue;
                const float v = shOut[w][lane];
                if (v > best) { best = v; bw = w; } // strict > : ties -> earliest
            }
            pick[s] = bw; pval[s] = best;
        }
        // restore temporal order (sort the 4 picks by index)
        #pragma unroll
        for (int i = 0; i < 3; ++i)
            #pragma unroll
            for (int j = 0; j < 3; ++j)
                if (pick[j] > pick[j + 1]) {
                    const int   ti = pick[j]; pick[j] = pick[j + 1]; pick[j + 1] = ti;
                    const float tv = pval[j]; pval[j] = pval[j + 1]; pval[j + 1] = tv;
                }
        float* outp = pooled2 + (size_t)b * FCDIM;
        #pragma unroll
        for (int s = 0; s < 4; ++s)
            outp[s * FEAT2 + lane * NG2 + gg] = tanhf(pval[s]);
    }
}

// ---------------------------------------------------------------------------
// Kernel 3: FC (N=1 GEMV) + [zeros, out] concat.  One block per batch row.
// ---------------------------------------------------------------------------
__global__ void __launch_bounds__(256)
dcnn_fc(const float* __restrict__ pooled2, const float* __restrict__ wfc,
        const float* __restrict__ bfc, float* __restrict__ out) {
    const int b = blockIdx.x;
    __shared__ float red[256];
    float s = 0.f;
    for (int f = threadIdx.x; f < FCDIM; f += 256)
        s += pooled2[(size_t)b * FCDIM + f] * wfc[f];
    red[threadIdx.x] = s;
    __syncthreads();
    for (int st = 128; st; st >>= 1) {
        if (threadIdx.x < st) red[threadIdx.x] += red[threadIdx.x + st];
        __syncthreads();
    }
    if (threadIdx.x == 0) {
        out[2 * b + 0] = 0.f;
        out[2 * b + 1] = red[0] + bfc[0];
    }
}

// ---------------------------------------------------------------------------
extern "C" void kernel_launch(void* const* d_in, const int* in_sizes, int n_in,
                              void* d_out, int out_size, void* d_ws, size_t ws_size,
                              hipStream_t stream) {
    const float* x       = (const float*)d_in[0];
    const int*   lengths = (const int*)  d_in[1];
    const float* w1      = (const float*)d_in[2];
    const float* b1      = (const float*)d_in[3];
    const float* w2      = (const float*)d_in[4];
    const float* b2      = (const float*)d_in[5];
    const float* wfc     = (const float*)d_in[6];
    const float* bfc     = (const float*)d_in[7];
    float*       out     = (float*)d_out;

    const int B = in_sizes[1];                  // 64

    // workspace layout: P1 (f16) | pooled2 (f32)
    _Float16* P1 = (_Float16*)d_ws;
    const size_t p1_bytes = (size_t)B * NG1 * 6 * K1MAX * sizeof(_Float16);
    float* pooled2 = (float*)((char*)d_ws + p1_bytes);

    (void)n_in; (void)out_size; (void)ws_size;

    dcnn_conv1_kmax<<<dim3(B * NG1), dim3(192), 0, stream>>>(
        x, lengths, w1, b1, P1);
    dcnn_conv2_kmax_wmma<<<dim3(B * NG2), dim3(32), 0, stream>>>(
        P1, lengths, w2, b2, pooled2);
    dcnn_fc<<<dim3(B), dim3(256), 0, stream>>>(
        pooled2, wfc, bfc, out);
}
